// MILPFAttnTrexModel_81647328297074
// MI455X (gfx1250) — matrix operand order repaired and checked by reading, not compiled, not run on hardware
//
#include <hip/hip_runtime.h>

// ---------------------------------------------------------------------------
// CDNA5 (gfx1250) wave32 WMMA implementation of the MIL-attention model.
// Matrix work runs on v_wmma_f32_16x16x32_bf16; the dominant GEMM stages its
// A-tile in LDS (conflict-free padded layout) so x is streamed from HBM once.
// ---------------------------------------------------------------------------

typedef __attribute__((ext_vector_type(16))) __bf16 v16bf;
typedef __attribute__((ext_vector_type(8)))  float  v8f;

union Frag16 {
    v16bf v;
    uint4 q[2];
    unsigned int u[8];
};

__device__ __forceinline__ unsigned short f2bf(float f) {
    __bf16 h = (__bf16)f;                    // RNE; lowers to v_cvt_*bf16_f32
    return __builtin_bit_cast(unsigned short, h);
}
__device__ __forceinline__ unsigned int pk2(float a, float b) {
    return (unsigned int)f2bf(a) | ((unsigned int)f2bf(b) << 16);
}
__device__ __forceinline__ float bf2f(unsigned short h) {
    unsigned int u = ((unsigned int)h) << 16;
    return __builtin_bit_cast(float, u);
}

// ---------------------------------------------------------------------------
// Generic bf16 WMMA GEMM:  C = act(A @ W + b)
//   A: [M,K] fp32 (converted on load) or bf16
//   WT: transposed weights, bf16, [Nc][K] row-major
//   block = 256 threads = 8 waves; wave w -> rows m0+w*16..+16,
//   cols n0..n0+NT*16 (NT accumulator fragments per wave).
// ---------------------------------------------------------------------------
template<int NT, bool A_F32, bool OUT_BF16, bool RELU>
__global__ __launch_bounds__(256) void gemm_wmma_kernel(
    const void* __restrict__ Ap,
    const unsigned short* __restrict__ WT,
    const float* __restrict__ bias,
    void* __restrict__ Cp,
    int M, int K, int Nc)
{
    const int tid  = threadIdx.x;
    const int wave = tid >> 5;
    const int lane = tid & 31;
    const int half = lane >> 4;   // 0: lanes 0-15, 1: lanes 16-31
    const int l16  = lane & 15;

    const int m0 = blockIdx.y * 128 + wave * 16;
    const int n0 = blockIdx.x * (NT * 16);

    int rowA = m0 + l16;                 // A frag: both lane halves hold M=0..15
    if (rowA > M - 1) rowA = M - 1;      // clamp OOB rows (results not stored)

    const float*          Af = (const float*)Ap;
    const unsigned short* Ah = (const unsigned short*)Ap;

    v8f acc[NT];
    #pragma unroll
    for (int t = 0; t < NT; ++t) acc[t] = (v8f){};

    for (int k0 = 0; k0 < K; k0 += 32) {
        // A fragment (16x32 bf16): lane half selects K offset (+0/+8);
        // VGPRs 0-3 -> K base, VGPRs 4-7 -> K base + 16.
        Frag16 a;
        if (A_F32) {
            const float* p = Af + (size_t)rowA * K + (k0 + half * 8);
            float4 f0 = *(const float4*)(p +  0);
            float4 f1 = *(const float4*)(p +  4);
            float4 f2 = *(const float4*)(p + 16);
            float4 f3 = *(const float4*)(p + 20);
            a.u[0] = pk2(f0.x, f0.y); a.u[1] = pk2(f0.z, f0.w);
            a.u[2] = pk2(f1.x, f1.y); a.u[3] = pk2(f1.z, f1.w);
            a.u[4] = pk2(f2.x, f2.y); a.u[5] = pk2(f2.z, f2.w);
            a.u[6] = pk2(f3.x, f3.y); a.u[7] = pk2(f3.z, f3.w);
        } else {
            const unsigned short* p = Ah + (size_t)rowA * K + (k0 + half * 8);
            a.q[0] = *(const uint4*)(p +  0);
            a.q[1] = *(const uint4*)(p + 16);
        }

        // B (32x16): lane's column contiguous in K in WT;
        // lane half selects K offset (+0/+8); VGPRs 4-7 -> K base + 16.
        #pragma unroll
        for (int t = 0; t < NT; ++t) {
            const int col = n0 + t * 16 + l16;
            const unsigned short* bp = WT + (size_t)col * K + (k0 + half * 8);
            Frag16 b;
            b.q[0] = *(const uint4*)(bp + 0);
            b.q[1] = *(const uint4*)(bp + 16);
            acc[t] = __builtin_amdgcn_wmma_f32_16x16x32_bf16(
                         false, a.v, false, b.v, (short)0, acc[t], false, false);
        }
    }

    // Epilogue. C/D layout: VGPR r, lane -> row = 8*half + r, col = l16.
    const bool full = (m0 + 16 <= M);
    #pragma unroll
    for (int t = 0; t < NT; ++t) {
        const int col = n0 + t * 16 + l16;
        const float bc = bias[col];
        #pragma unroll
        for (int r = 0; r < 8; ++r) {
            const int row = m0 + half * 8 + r;
            if (!full && row >= M) continue;
            float xv = acc[t][r] + bc;
            if (RELU) xv = fmaxf(xv, 0.f);
            if (OUT_BF16)
                ((unsigned short*)Cp)[(size_t)row * Nc + col] = f2bf(xv);
            else
                ((float*)Cp)[(size_t)row * Nc + col] = xv;
        }
    }
}

// ---------------------------------------------------------------------------
// GEMM1 (x fp32 [M,1024] @ Wl0 -> relu -> h1 bf16 [M,512]).
// Stages one 64-row A tile in LDS as bf16 (row stride 1032 -> conflict-free
// ds_load_b128 fragments) and iterates over all 512 columns in-kernel, so the
// 268 MB x stream is read from HBM exactly once.
// 8 waves = 4 row-waves x 2 col-waves; col loop runs 4 x 128 columns.
// ---------------------------------------------------------------------------
__global__ __launch_bounds__(256) void gemm1_lds_kernel(
    const float* __restrict__ X,
    const unsigned short* __restrict__ WT,
    const float* __restrict__ bias,
    unsigned short* __restrict__ Out,
    int M)
{
    constexpr int K = 1024, Nc = 512, LDK = 1032;  // 8-elem (16B) row padding
    extern __shared__ unsigned short sA[];         // 64 * 1032 * 2 = 132,096 B

    const int tid  = threadIdx.x;
    const int wave = tid >> 5;
    const int lane = tid & 31;
    const int half = lane >> 4;
    const int l16  = lane & 15;
    const int rwave = wave & 3;   // row tile 0..3
    const int cwave = wave >> 2;  // col half 0..1
    const int blockRow = blockIdx.x * 64;

    // ---- cooperative fill: 64 rows x 1024 cols, fp32 -> bf16, coalesced
    for (int i = tid; i < 64 * 256; i += 256) {
        const int row = i >> 8;
        const int k4  = (i & 255) * 4;
        int grow = blockRow + row;
        if (grow > M - 1) grow = M - 1;
        float4 f = *(const float4*)(X + (size_t)grow * K + k4);
        uint2 z;
        z.x = pk2(f.x, f.y);
        z.y = pk2(f.z, f.w);
        *(uint2*)(sA + row * LDK + k4) = z;
    }
    __syncthreads();

    const unsigned short* aBase = sA + (rwave * 16 + l16) * LDK + half * 8;

    for (int cg = 0; cg < 4; ++cg) {
        const int colBase = cg * 128 + cwave * 64;
        v8f acc[4];
        #pragma unroll
        for (int t = 0; t < 4; ++t) acc[t] = (v8f){};

        for (int k0 = 0; k0 < K; k0 += 32) {
            Frag16 a;
            const unsigned short* ap = aBase + k0;
            a.q[0] = *(const uint4*)(ap +  0);   // ds_load_b128
            a.q[1] = *(const uint4*)(ap + 16);
            #pragma unroll
            for (int t = 0; t < 4; ++t) {
                const int col = colBase + t * 16 + l16;
                const unsigned short* bp = WT + (size_t)col * K + (k0 + half * 8);
                Frag16 b;
                b.q[0] = *(const uint4*)(bp + 0);
                b.q[1] = *(const uint4*)(bp + 16);
                acc[t] = __builtin_amdgcn_wmma_f32_16x16x32_bf16(
                             false, a.v, false, b.v, (short)0, acc[t], false, false);
            }
        }

        const int rowTop = blockRow + rwave * 16 + half * 8;
        const bool full = (blockRow + 64 <= M);
        #pragma unroll
        for (int t = 0; t < 4; ++t) {
            const int col = colBase + t * 16 + l16;
            const float bc = bias[col];
            #pragma unroll
            for (int r = 0; r < 8; ++r) {
                const int row = rowTop + r;
                if (!full && row >= M) continue;
                float xv = fmaxf(acc[t][r] + bc, 0.f);
                Out[(size_t)row * Nc + col] = f2bf(xv);
            }
        }
    }
}

// ---------------------------------------------------------------------------
// Weight transpose + fp32->bf16:  WT[n*K + k] = bf16(W[k*Nc + n])
// ---------------------------------------------------------------------------
__global__ __launch_bounds__(256) void tconv_kernel(
    const float* __restrict__ W, unsigned short* __restrict__ WT, int K, int Nc)
{
    long idx = (long)blockIdx.x * 256 + threadIdx.x;
    long total = (long)K * Nc;
    if (idx >= total) return;
    int n = (int)(idx / K);
    int k = (int)(idx % K);
    WT[idx] = f2bf(W[(size_t)k * Nc + n]);
}

// ---------------------------------------------------------------------------
// Fold:  Wk2[l][c] = (1/16) sum_n Wk[c][n]*latent[l][n];  bk2[l] likewise.
// ---------------------------------------------------------------------------
__global__ __launch_bounds__(256) void fold_kernel(
    const float* __restrict__ Wk, const float* __restrict__ bk,
    const float* __restrict__ latent,
    float* __restrict__ Wk2, float* __restrict__ bk2)
{
    int idx = blockIdx.x * 256 + threadIdx.x;
    const float scale = 0.0625f;  // 1/sqrt(256)
    if (idx < 8 * 256) {
        int l = idx >> 8, c = idx & 255;
        float s = 0.f;
        for (int n = 0; n < 256; ++n)
            s += Wk[(size_t)c * 256 + n] * latent[l * 256 + n];
        Wk2[l * 256 + c] = s * scale;
    } else if (idx < 8 * 256 + 8) {
        int l = idx - 2048;
        float s = 0.f;
        for (int n = 0; n < 256; ++n) s += bk[n] * latent[l * 256 + n];
        bk2[l] = s * scale;
    }
}

// ---------------------------------------------------------------------------
// scores[r][l] = bk2[l] + sum_c h2[r][c] * Wk2[l][c]   (thin GEMV, L=8)
// ---------------------------------------------------------------------------
__global__ __launch_bounds__(256) void scores_kernel(
    const unsigned short* __restrict__ h2,
    const float* __restrict__ Wk2, const float* __restrict__ bk2,
    float* __restrict__ sc, int Nrows)
{
    int r = blockIdx.x * 256 + threadIdx.x;
    if (r >= Nrows) return;
    float acc[8];
    #pragma unroll
    for (int l = 0; l < 8; ++l) acc[l] = bk2[l];
    const unsigned short* hp = h2 + (size_t)r * 256;
    for (int c = 0; c < 256; c += 8) {
        uint4 q = *(const uint4*)(hp + c);
        float hv[8];
        hv[0] = bf2f((unsigned short)(q.x));
        hv[1] = bf2f((unsigned short)(q.x >> 16));
        hv[2] = bf2f((unsigned short)(q.y));
        hv[3] = bf2f((unsigned short)(q.y >> 16));
        hv[4] = bf2f((unsigned short)(q.z));
        hv[5] = bf2f((unsigned short)(q.z >> 16));
        hv[6] = bf2f((unsigned short)(q.w));
        hv[7] = bf2f((unsigned short)(q.w >> 16));
        #pragma unroll
        for (int j = 0; j < 8; ++j) {
            #pragma unroll
            for (int l = 0; l < 8; ++l)
                acc[l] += hv[j] * Wk2[l * 256 + c + j];  // uniform -> scalar loads
        }
    }
    #pragma unroll
    for (int l = 0; l < 8; ++l) sc[(size_t)r * 8 + l] = acc[l];
}

// ---------------------------------------------------------------------------
// Per-group segment softmax + weighted sum. Deterministic: order-preserving
// compacted row list per group, fixed accumulation order (no float atomics).
// out_group[g][l*256 + d] = (sum_r e_l * v[r][d]) / (sum_r e_l)
// ---------------------------------------------------------------------------
#define ATTN_MAXL 4096

__global__ __launch_bounds__(256) void attn_kernel(
    const int* __restrict__ grp, const int* __restrict__ itype,
    const float* __restrict__ sc, const float* __restrict__ vbuf,
    int* __restrict__ lists, float* __restrict__ out_group, int Nrows)
{
    __shared__ int   s_cnt[256];
    __shared__ int   s_off[256];
    __shared__ float s_m[256 * 8];
    __shared__ float s_max[8];
    __shared__ int   s_total;

    const int g = blockIdx.x;
    const int t = threadIdx.x;
    const int CH = (Nrows + 255) / 256;
    const int r0 = t * CH;
    const int r1 = (r0 + CH < Nrows) ? (r0 + CH) : Nrows;

    float mx[8];
    #pragma unroll
    for (int l = 0; l < 8; ++l) mx[l] = -INFINITY;
    int cnt = 0;
    for (int r = r0; r < r1; ++r) {
        if (itype[r] == 1 && grp[r] == g) {
            ++cnt;
            const float* sp = sc + (size_t)r * 8;
            #pragma unroll
            for (int l = 0; l < 8; ++l) mx[l] = fmaxf(mx[l], sp[l]);
        }
    }
    s_cnt[t] = cnt;
    #pragma unroll
    for (int l = 0; l < 8; ++l) s_m[t * 8 + l] = mx[l];
    __syncthreads();

    if (t == 0) {
        int o = 0;
        for (int i = 0; i < 256; ++i) { s_off[i] = o; o += s_cnt[i]; }
        s_total = o;
    }
    if (t < 8) {
        float m = -INFINITY;
        for (int i = 0; i < 256; ++i) m = fmaxf(m, s_m[i * 8 + t]);
        s_max[t] = m;
    }
    __syncthreads();

    int* mylist = lists + (size_t)g * ATTN_MAXL;
    int w = s_off[t];
    for (int r = r0; r < r1; ++r) {
        if (itype[r] == 1 && grp[r] == g) {
            if (w < ATTN_MAXL) mylist[w] = r;
            ++w;
        }
    }
    __threadfence_block();
    __syncthreads();

    const int total = (s_total < ATTN_MAXL) ? s_total : ATTN_MAXL;
    float smax[8];
    #pragma unroll
    for (int l = 0; l < 8; ++l) smax[l] = s_max[l];

    float den[8], acc[8];
    #pragma unroll
    for (int l = 0; l < 8; ++l) { den[l] = 0.f; acc[l] = 0.f; }
    for (int i = 0; i < total; ++i) {
        const int r = mylist[i];
        const float* sp = sc + (size_t)r * 8;
        const float  vv = vbuf[(size_t)r * 256 + t];
        #pragma unroll
        for (int l = 0; l < 8; ++l) {
            float e = __expf(sp[l] - smax[l]);
            den[l] += e;
            acc[l] += e * vv;
        }
    }
    #pragma unroll
    for (int l = 0; l < 8; ++l)
        out_group[(size_t)g * 2048 + l * 256 + t] = acc[l] / den[l];
}

// ---------------------------------------------------------------------------
// Final:  out[g][c] = bout[c] + fused[g] . Wout[:,c]   (fused = [whole|attn])
// ---------------------------------------------------------------------------
__global__ __launch_bounds__(256) void final_kernel(
    const float* __restrict__ whole, const float* __restrict__ og,
    const float* __restrict__ Wout, const float* __restrict__ bout,
    float* __restrict__ out)
{
    __shared__ float red[256 * 2];
    const int g = blockIdx.x;
    const int t = threadIdx.x;
    float a0 = 0.f, a1 = 0.f;
    for (int j = t; j < 2560; j += 256) {
        float f = (j < 512) ? whole[(size_t)g * 512 + j]
                            : og[(size_t)g * 2048 + (j - 512)];
        a0 += f * Wout[j * 2 + 0];
        a1 += f * Wout[j * 2 + 1];
    }
    red[t * 2 + 0] = a0;
    red[t * 2 + 1] = a1;
    __syncthreads();
    for (int s = 128; s > 0; s >>= 1) {
        if (t < s) {
            red[t * 2 + 0] += red[(t + s) * 2 + 0];
            red[t * 2 + 1] += red[(t + s) * 2 + 1];
        }
        __syncthreads();
    }
    if (t == 0) {
        out[g * 2 + 0] = red[0] + bout[0];
        out[g * 2 + 1] = red[1] + bout[1];
    }
}

// ---------------------------------------------------------------------------
extern "C" void kernel_launch(void* const* d_in, const int* in_sizes, int n_in,
                              void* d_out, int out_size, void* d_ws, size_t ws_size,
                              hipStream_t stream)
{
    (void)n_in; (void)out_size; (void)ws_size;

    constexpr int IN = 1024, GH = 1024, GL = 512, LH = 512, LC = 256, L = 8;
    constexpr int G = 64;

    const float* x    = (const float*)d_in[0];
    const int*   grp  = (const int*)d_in[1];
    const int*   ity  = (const int*)d_in[2];
    const float* Wg0  = (const float*)d_in[3];
    const float* bg0  = (const float*)d_in[4];
    const float* Wg1  = (const float*)d_in[5];
    const float* bg1  = (const float*)d_in[6];
    const float* Wl0  = (const float*)d_in[7];
    const float* bl0  = (const float*)d_in[8];
    const float* Wl1  = (const float*)d_in[9];
    const float* bl1  = (const float*)d_in[10];
    const float* Wk   = (const float*)d_in[11];
    const float* bk   = (const float*)d_in[12];
    const float* Wv   = (const float*)d_in[13];
    const float* bv   = (const float*)d_in[14];
    const float* lat  = (const float*)d_in[15];
    const float* Wout = (const float*)d_in[16];
    const float* bout = (const float*)d_in[17];
    float* out = (float*)d_out;

    const int Nrows = in_sizes[1];  // 65600

    char* ws = (char*)d_ws;
    size_t off = 0;
    auto alloc = [&](size_t bytes) -> void* {
        void* p = ws + off;
        off = (off + bytes + 255) & ~(size_t)255;
        return p;
    };
    unsigned short* wtg0 = (unsigned short*)alloc((size_t)GH * IN * 2);
    unsigned short* wtg1 = (unsigned short*)alloc((size_t)GL * GH * 2);
    unsigned short* wtl0 = (unsigned short*)alloc((size_t)LH * IN * 2);
    unsigned short* wtl1 = (unsigned short*)alloc((size_t)LC * LH * 2);
    unsigned short* wtv  = (unsigned short*)alloc((size_t)LC * LC * 2);
    float* wk2  = (float*)alloc((size_t)L * LC * 4);
    float* bk2  = (float*)alloc(256);
    unsigned short* wh1 = (unsigned short*)alloc((size_t)G * GH * 2);
    float* wagg = (float*)alloc((size_t)G * GL * 4);
    unsigned short* h1 = (unsigned short*)alloc((size_t)Nrows * LH * 2);
    unsigned short* h2 = (unsigned short*)alloc((size_t)Nrows * LC * 2);
    float* vbuf = (float*)alloc((size_t)Nrows * LC * 4);
    float* sc   = (float*)alloc((size_t)Nrows * L * 4);
    int*   lists = (int*)alloc((size_t)G * ATTN_MAXL * 4);
    float* og   = (float*)alloc((size_t)G * L * LC * 4);

    // 1) weight transpose + bf16 convert
    tconv_kernel<<<(GH * IN + 255) / 256, 256, 0, stream>>>(Wg0, wtg0, IN, GH);
    tconv_kernel<<<(GL * GH + 255) / 256, 256, 0, stream>>>(Wg1, wtg1, GH, GL);
    tconv_kernel<<<(LH * IN + 255) / 256, 256, 0, stream>>>(Wl0, wtl0, IN, LH);
    tconv_kernel<<<(LC * LH + 255) / 256, 256, 0, stream>>>(Wl1, wtl1, LH, LC);
    tconv_kernel<<<(LC * LC + 255) / 256, 256, 0, stream>>>(Wv,  wtv,  LC, LC);

    // 2) fold Wk/latent -> thin score projection
    fold_kernel<<<9, 256, 0, stream>>>(Wk, bk, lat, wk2, bk2);

    // 3) whole-image branch (only first 64 rows matter due to segment_max)
    gemm_wmma_kernel<8, true,  true,  true><<<dim3(GH / 128, 1), 256, 0, stream>>>(
        x, wtg0, bg0, wh1, G, IN, GH);
    gemm_wmma_kernel<8, false, false, true><<<dim3(GL / 128, 1), 256, 0, stream>>>(
        wh1, wtg1, bg1, wagg, G, GH, GL);

    // 4) tile branch MLP + v projection (bf16 WMMA)
    //    GEMM1: LDS-staged A, x streamed from HBM exactly once.
    const int gb = (Nrows + 63) / 64;
    gemm1_lds_kernel<<<gb, 256, 64 * 1032 * 2, stream>>>(x, wtl0, bl0, h1, Nrows);

    const int gy = (Nrows + 127) / 128;
    gemm_wmma_kernel<8, false, true,  true ><<<dim3(LC / 128, gy), 256, 0, stream>>>(
        h1, wtl1, bl1, h2, Nrows, LH, LC);
    gemm_wmma_kernel<8, false, false, false><<<dim3(LC / 128, gy), 256, 0, stream>>>(
        h2, wtv, bv, vbuf, Nrows, LC, LC);

    // 5) scores, 6) deterministic segment softmax + weighted sum, 7) head
    scores_kernel<<<(Nrows + 255) / 256, 256, 0, stream>>>(h2, wk2, bk2, sc, Nrows);
    attn_kernel<<<G, 256, 0, stream>>>(grp, ity, sc, vbuf, lists, og, Nrows);
    final_kernel<<<G, 256, 0, stream>>>(wagg, og, Wout, bout, out);
}
